// SSCA_47107201302628
// MI455X (gfx1250) — compile-verified, matrix operand order checked by reading, not run on hardware
//
#include <hip/hip_runtime.h>
#include <hip/hip_bf16.h>
#include <math.h>

// ---- problem constants (match reference) ----
#define B0 32
#define C0 512
#define H0 64
#define W0 64
#define DC 32
#define NS 16
#define RK 32
#define L0 512           // sequence length == C0
#define PLANE (H0*W0)    // 4096

typedef __attribute__((ext_vector_type(16))) _Float16 v16h;
typedef __attribute__((ext_vector_type(8)))  float    v8f;

// K-index of element j in a 16-bit A (16x32) / B (32x16) WMMA fragment (wave32):
// lanes 0-15:  VGPR0..3 -> K 0..7,  VGPR4..7 -> K 16..23
// lanes 16-31: VGPR0..3 -> K 8..15, VGPR4..7 -> K 24..31
__device__ __forceinline__ int frag_k(int lane, int j) {
    int b = (j < 8) ? j : (j + 8);
    return (lane < 16) ? b : (b + 8);
}

__device__ __forceinline__ float gelu_exact(float v) {
    return 0.5f * v * (1.0f + erff(v * 0.70710678118654752f));
}

// ---------------------------------------------------------------------------
// Kernel 1: per-(b,c) plane mean + max.  grid = B0*C0, block = 256.
// ---------------------------------------------------------------------------
__global__ __launch_bounds__(256)
void ssca_reduce(const float* __restrict__ x,
                 float* __restrict__ avgp, float* __restrict__ mxp) {
    const int bc = blockIdx.x;
    const float4* xp = (const float4*)(x + (size_t)bc * PLANE);
    float s = 0.0f, m = -INFINITY;
#pragma unroll
    for (int i = 0; i < 4; ++i) {
        float4 v = xp[i * 256 + threadIdx.x];
        s += (v.x + v.y) + (v.z + v.w);
        m = fmaxf(m, fmaxf(fmaxf(v.x, v.y), fmaxf(v.z, v.w)));
    }
#pragma unroll
    for (int off = 16; off >= 1; off >>= 1) {
        s += __shfl_xor(s, off, 32);
        m = fmaxf(m, __shfl_xor(m, off, 32));
    }
    __shared__ float rs[8], rm[8];
    const int wid = threadIdx.x >> 5, lane = threadIdx.x & 31;
    if (lane == 0) { rs[wid] = s; rm[wid] = m; }
    __syncthreads();
    if (threadIdx.x == 0) {
        float ts = 0.0f, tm = -INFINITY;
#pragma unroll
        for (int i = 0; i < 8; ++i) { ts += rs[i]; tm = fmaxf(tm, rm[i]); }
        avgp[bc] = ts * (1.0f / (float)PLANE);
        mxp[bc]  = tm;
    }
}

// ---------------------------------------------------------------------------
// Kernel 2: whole middle stage for one batch.  grid = B0, block = 512 (16 waves)
// ---------------------------------------------------------------------------
__global__ __launch_bounds__(512)
void ssca_mid(const float* __restrict__ avgp, const float* __restrict__ mxp,
              const float* __restrict__ xc_w,    // (2,64,32)
              const float* __restrict__ dtc_w,   // (2,32,32)
              const float* __restrict__ dtc_b,   // (2,32)
              const float* __restrict__ Alog,    // (64,16)
              const float* __restrict__ Dcs,     // (64)
              const float* __restrict__ w_cin,   // (32,2)
              const float* __restrict__ bng, const float* __restrict__ bnb,
              const float* __restrict__ bnm, const float* __restrict__ bnv,
              const float* __restrict__ w_cout,  // (32)
              const float* __restrict__ lng, const float* __restrict__ lnb, // (512)
              float* __restrict__ attn) {        // (B0,512)
    __shared__ float h_s[DC * L0];      // 64 KB : h[d][l] (post BN+GELU)
    __shared__ float work_s[DC * L0];   // 64 KB : dts_in then delta (softplus)
    __shared__ float Bst[L0 * NS];      // 32 KB : B[l][n]
    __shared__ float Cst[L0 * NS];      // 32 KB : C[l][n]
    __shared__ float yacc[L0];          //  2 KB
    __shared__ float red[32];
    __shared__ float stats[2];

    const int b    = blockIdx.x;
    const int t    = threadIdx.x;
    const int lane = t & 31;
    const int wid  = t >> 5;

    // ---- phase 1: h[d][l] = GELU(BN(avg*w0 + mx*w1)) ----
    {
        const float a = avgp[b * L0 + t];
        const float m = mxp[b * L0 + t];
#pragma unroll 4
        for (int d = 0; d < DC; ++d) {
            float v = a * w_cin[2 * d] + m * w_cin[2 * d + 1];
            v = (v - bnm[d]) * rsqrtf(bnv[d] + 1e-5f);
            v = v * bng[d] + bnb[d];
            h_s[d * L0 + t] = gelu_exact(v);
        }
        yacc[t] = 0.0f;
    }
    __syncthreads();

    for (int k = 0; k < 2; ++k) {
        // ---- GEMM1: dbl[c,l] = sum_d xc_w[k][c][d] * xk[d,l]  (64x512x32)
        // wave `wid` owns ntiles {2*wid, 2*wid+1}, all 4 mtiles.
        v16h afr[4];
#pragma unroll
        for (int mt = 0; mt < 4; ++mt) {
            const int c = mt * 16 + (lane & 15);
#pragma unroll
            for (int j = 0; j < 16; ++j)
                afr[mt][j] = (_Float16)xc_w[(k * 64 + c) * DC + frag_k(lane, j)];
        }
#pragma unroll
        for (int ni = 0; ni < 2; ++ni) {
            const int l    = (wid * 2 + ni) * 16 + (lane & 15);   // scan-domain col
            const int lcol = k ? (L0 - 1 - l) : l;                // flip for k=1
            v16h bfr;
#pragma unroll
            for (int j = 0; j < 16; ++j)
                bfr[j] = (_Float16)h_s[frag_k(lane, j) * L0 + lcol];
#pragma unroll
            for (int mt = 0; mt < 4; ++mt) {
                v8f acc = {};
                acc = __builtin_amdgcn_wmma_f32_16x16x32_f16(
                    false, afr[mt], false, bfr, (short)0, acc, false, false);
#pragma unroll
                for (int r = 0; r < 8; ++r) {
                    const int c = mt * 16 + r + ((lane < 16) ? 0 : 8);
                    const float v = acc[r];
                    if (c < RK)            work_s[c * L0 + l]      = v;  // dts
                    else if (c < RK + NS)  Bst[l * NS + (c - RK)]  = v;  // Bs
                    else                   Cst[l * NS + (c - RK - NS)] = v; // Cs
                }
            }
        }
        __syncthreads();

        // ---- GEMM2: delta[d,l] = softplus(sum_r dtc_w[k][d][r]*dts[r,l] + bias)
        v16h b2[2];
#pragma unroll
        for (int ni = 0; ni < 2; ++ni) {
            const int l = (wid * 2 + ni) * 16 + (lane & 15);
#pragma unroll
            for (int j = 0; j < 16; ++j)
                b2[ni][j] = (_Float16)work_s[frag_k(lane, j) * L0 + l];
        }
        v16h a2[2];
#pragma unroll
        for (int mt = 0; mt < 2; ++mt) {
            const int d = mt * 16 + (lane & 15);
#pragma unroll
            for (int j = 0; j < 16; ++j)
                a2[mt][j] = (_Float16)dtc_w[(k * DC + d) * RK + frag_k(lane, j)];
        }
        __syncthreads();   // everyone captured their B frags; safe to overwrite work_s
#pragma unroll
        for (int ni = 0; ni < 2; ++ni) {
            const int l = (wid * 2 + ni) * 16 + (lane & 15);
#pragma unroll
            for (int mt = 0; mt < 2; ++mt) {
                v8f acc = {};
                acc = __builtin_amdgcn_wmma_f32_16x16x32_f16(
                    false, a2[mt], false, b2[ni], (short)0, acc, false, false);
#pragma unroll
                for (int r = 0; r < 8; ++r) {
                    const int d  = mt * 16 + r + ((lane < 16) ? 0 : 8);
                    const float dv = acc[r] + dtc_b[k * DC + d];
                    work_s[d * L0 + l] = (dv > 20.0f) ? dv : log1pf(__expf(dv));
                }
            }
        }
        __syncthreads();

        // ---- selective scan: thread t = state (d = t>>4, n = t&15) ----
        {
            const int d   = t >> 4;
            const int n   = t & 15;
            const int row = k * DC + d;
            const float And = -__expf(Alog[row * NS + n]);
            const float Dd  = Dcs[row];
            const float wc  = w_cout[d];
            float hs = 0.0f;
            for (int l = 0; l < L0; ++l) {
                const float delta = work_s[d * L0 + l];
                const float u  = h_s[d * L0 + (k ? (L0 - 1 - l) : l)];
                const float Bv = Bst[l * NS + n];
                const float Cv = Cst[l * NS + n];
                hs = __expf(delta * And) * hs + delta * Bv * u;
                float v = hs * Cv;
                v += __shfl_xor(v, 8, 32);
                v += __shfl_xor(v, 4, 32);
                v += __shfl_xor(v, 2, 32);
                v += __shfl_xor(v, 1, 32);
                if (n == 0)
                    atomicAdd(&yacc[k ? (L0 - 1 - l) : l], (v + Dd * u) * wc);
            }
        }
        __syncthreads();
    }

    // ---- phase 4: y -> GELU -> LayerNorm over L0 -> attn ----
    const float yg = gelu_exact(yacc[t]);
    float s = yg, s2 = yg * yg;
#pragma unroll
    for (int off = 16; off >= 1; off >>= 1) {
        s  += __shfl_xor(s,  off, 32);
        s2 += __shfl_xor(s2, off, 32);
    }
    if (lane == 0) { red[wid] = s; red[wid + 16] = s2; }
    __syncthreads();
    if (t == 0) {
        float ts = 0.0f, ts2 = 0.0f;
#pragma unroll
        for (int i = 0; i < 16; ++i) { ts += red[i]; ts2 += red[i + 16]; }
        const float mu  = ts * (1.0f / (float)L0);
        const float var = ts2 * (1.0f / (float)L0) - mu * mu;
        stats[0] = mu;
        stats[1] = rsqrtf(var + 1e-5f);
    }
    __syncthreads();
    attn[b * L0 + t] = (yg - stats[0]) * stats[1] * lng[t] + lnb[t];
}

// ---------------------------------------------------------------------------
// Kernel 3: out = x * (1 + attn[b,c]).  grid = B0*C0, block = 256.
// ---------------------------------------------------------------------------
__global__ __launch_bounds__(256)
void ssca_apply(const float* __restrict__ x, const float* __restrict__ attn,
                float* __restrict__ out) {
    const int bc = blockIdx.x;
    const float scale = 1.0f + attn[bc];
    const float4* xp = (const float4*)(x   + (size_t)bc * PLANE);
    float4*       op = (float4*)      (out + (size_t)bc * PLANE);
#pragma unroll
    for (int i = 0; i < 4; ++i) {
        float4 v = xp[i * 256 + threadIdx.x];
        v.x *= scale; v.y *= scale; v.z *= scale; v.w *= scale;
        op[i * 256 + threadIdx.x] = v;
    }
}

extern "C" void kernel_launch(void* const* d_in, const int* in_sizes, int n_in,
                              void* d_out, int out_size, void* d_ws, size_t ws_size,
                              hipStream_t stream) {
    const float* x      = (const float*)d_in[0];
    const float* xc_w   = (const float*)d_in[1];
    const float* dtc_w  = (const float*)d_in[2];
    const float* dtc_b  = (const float*)d_in[3];
    const float* Alog   = (const float*)d_in[4];
    const float* Dcs    = (const float*)d_in[5];
    const float* w_cin  = (const float*)d_in[6];
    const float* bng    = (const float*)d_in[7];
    const float* bnb    = (const float*)d_in[8];
    const float* bnm    = (const float*)d_in[9];
    const float* bnv    = (const float*)d_in[10];
    const float* w_cout = (const float*)d_in[11];
    const float* lng    = (const float*)d_in[12];
    const float* lnb    = (const float*)d_in[13];

    float* ws   = (float*)d_ws;
    float* avgp = ws;                    // B0*C0
    float* mxp  = ws + B0 * C0;          // B0*C0
    float* attn = ws + 2 * B0 * C0;      // B0*C0

    ssca_reduce<<<B0 * C0, 256, 0, stream>>>(x, avgp, mxp);
    ssca_mid<<<B0, 512, 0, stream>>>(avgp, mxp, xc_w, dtc_w, dtc_b, Alog, Dcs,
                                     w_cin, bng, bnb, bnm, bnv, w_cout, lng, lnb,
                                     attn);
    ssca_apply<<<B0 * C0, 256, 0, stream>>>(x, attn, (float*)d_out);
}
